// GPT_64192581206199
// MI455X (gfx1250) — compile-verified, matrix operand order checked by reading, not run on hardware
//
#include <hip/hip_runtime.h>

// ---------------------------------------------------------------------------
// GPT forward for MI455X (gfx1250): bf16 WMMA everywhere, f32 residual stream.
// GEMM stages weight slabs into LDS via async global->LDS copies (ASYNCcnt),
// double-buffered, shared by all 4 waves of the block.
// ---------------------------------------------------------------------------

typedef __bf16 bf16;
typedef bf16 v16bf __attribute__((ext_vector_type(16)));
typedef bf16 v8bf  __attribute__((ext_vector_type(8)));
typedef float v8f  __attribute__((ext_vector_type(8)));
typedef int  v4i   __attribute__((vector_size(16)));

#define DEVI __device__ __forceinline__

constexpr int L_BLOCKS = 12;
constexpr int EMB   = 1024;
constexpr int HEADS = 16;
constexpr int HS    = 64;
constexpr int VOCAB = 32000;
constexpr int BATCH = 2;
constexpr int SEQ   = 2048;
constexpr int NTOK  = BATCH * SEQ;   // 4096

DEVI v8f wmma_bf16(v16bf a, v16bf b, v8f c) {
  // D(16x16,f32) = A(16x32,bf16) * B(32x16,bf16) + C
  return __builtin_amdgcn_wmma_f32_16x16x32_bf16(
      /*neg_a=*/false, a, /*neg_b=*/false, b,
      /*c_mod=*/(short)0, c, /*reuse_a=*/false, /*reuse_b=*/false);
}

// Load one 16-element half-wave fragment slice: p points at (row, kb+koff).
// Per the 16-bit A/B VGPR layout, a lane holds K = {koff..koff+7} and
// {koff+16..koff+23} of its row/column, each an aligned 16-byte chunk.
DEVI v16bf load_frag(const bf16* p) {
  v8bf lo = *(const v8bf*)p;
  v8bf hi = *(const v8bf*)(p + 16);
  v16bf f;
#pragma unroll
  for (int i = 0; i < 8; ++i) { f[i] = lo[i]; f[i + 8] = hi[i]; }
  return f;
}

DEVI float gelu_exact(float x) {
  return 0.5f * x * (1.0f + erff(x * 0.70710678118654752f));
}

// ---- async global->LDS copy (GLOBAL_LOAD_ASYNC_TO_LDS_B128, ASYNCcnt) -----
#if __has_builtin(__builtin_amdgcn_global_load_async_to_lds_b128)
#define HAVE_ASYNC_LDS 1
typedef __attribute__((address_space(1))) v4i* as1_v4i_ptr;  // global int4*
typedef __attribute__((address_space(3))) v4i* as3_v4i_ptr;  // LDS int4*
DEVI void async_copy16(const bf16* g, bf16* l) {
  v4i* gg = (v4i*)const_cast<bf16*>(g);
  v4i* ll = (v4i*)l;
  __builtin_amdgcn_global_load_async_to_lds_b128(
      (as1_v4i_ptr)gg, (as3_v4i_ptr)ll, /*offset=*/0, /*cpol=*/0);
}
#else
#define HAVE_ASYNC_LDS 0
DEVI void async_copy16(const bf16* g, bf16* l) {
  *(v8bf*)l = *(const v8bf*)g;   // synchronous fallback
}
#endif

DEVI void wait_async_le(int n) {
#if HAVE_ASYNC_LDS
#if __has_builtin(__builtin_amdgcn_s_wait_asynccnt)
  if (n == 0) __builtin_amdgcn_s_wait_asynccnt(0);
  else        __builtin_amdgcn_s_wait_asynccnt(2);
#else
  if (n == 0) asm volatile("s_wait_asynccnt 0x0" ::: "memory");
  else        asm volatile("s_wait_asynccnt 0x2" ::: "memory");
#endif
#else
  (void)n;
#endif
}

// ---------------------------------------------------------------------------
// Generic WMMA GEMM: C[M,N] = A[M,K](bf16) * W[N,K](bf16)^T (+bias,+gelu,+resid)
// Block = 4 waves stacked in M (block tile 256x64). The 64x32 W slab for the
// block's 64 output columns is staged in LDS (async copy, double buffered)
// and shared by all waves. Each wave computes a 64x64 tile (4x4 WMMAs).
// blockIdx.z batches over heads for the QKV projections.
// EPI bits: 1=bf16 out, 2=bias, 4=gelu, 8=residual(f32), 16=transposed V out.
// Launch tiling must be exact (block barriers inside).
// ---------------------------------------------------------------------------
template <int EPI>
__global__ __launch_bounds__(128)
void gemm_ker(const bf16* __restrict__ A, int lda,
              const bf16* __restrict__ W, int ldw,
              const float* __restrict__ bias,
              float* __restrict__ Cf, bf16* __restrict__ Cb, int ldc,
              const float* __restrict__ resid,
              int M, int N, int K,
              long a_zoff, long w_zoff, long c_zoff)
{
  constexpr bool OUT_BF16 = (EPI & 1) != 0;
  constexpr bool HAS_BIAS = (EPI & 2) != 0;
  constexpr bool DO_GELU  = (EPI & 4) != 0;
  constexpr bool DO_RESID = (EPI & 8) != 0;
  constexpr bool TRANS_V  = (EPI & 16) != 0;

  __shared__ __align__(16) bf16 wtile[2][64 * 32];   // 8 KB, double buffered

  const int z = blockIdx.z;
  A += (long)z * a_zoff;
  W += (long)z * w_zoff;
  if (OUT_BF16 && !TRANS_V) Cb += (long)z * c_zoff;

  const int wave = threadIdx.x >> 5;
  const int lane = threadIdx.x & 31;
  const int ln   = lane & 15;
  const int hf   = lane >> 4;       // half-wave select
  const int koff = hf * 8;          // K interleave per 16-bit WMMA layout

  const int m0 = (blockIdx.x * 4 + wave) * 64;
  const int n0 = blockIdx.y * 64;

  // W-slab staging: 64 rows x 32 K bf16 = 4 KB = 256 16-byte chunks;
  // 128 threads move 2 chunks each. chunk c -> row c>>2, sub c&3.
  const int c0 = threadIdx.x * 2;
  const int r0 = c0 >> 2,        q0 = (c0 & 3) * 8;
  const int r1 = (c0 + 1) >> 2,  q1 = ((c0 + 1) & 3) * 8;

  {
    const int kb = 0;
    async_copy16(W + (long)(n0 + r0) * ldw + kb + q0, &wtile[0][r0 * 32 + q0]);
    async_copy16(W + (long)(n0 + r1) * ldw + kb + q1, &wtile[0][r1 * 32 + q1]);
  }

  v8f acc[4][4] = {};

  for (int kb = 0; kb < K; kb += 32) {
    const int cur = (kb >> 5) & 1;
    const bool has_next = (kb + 32) < K;
    if (has_next) {
      const int nb = kb + 32;
      async_copy16(W + (long)(n0 + r0) * ldw + nb + q0, &wtile[cur ^ 1][r0 * 32 + q0]);
      async_copy16(W + (long)(n0 + r1) * ldw + nb + q1, &wtile[cur ^ 1][r1 * 32 + q1]);
    }
    wait_async_le(has_next ? 2 : 0);   // current slab's copies complete
    __syncthreads();                   // ... in every wave of the block

    v16bf af[4], wf[4];
#pragma unroll
    for (int i = 0; i < 4; ++i)
      af[i] = load_frag(A + (long)(m0 + i * 16 + ln) * lda + kb + koff);
#pragma unroll
    for (int j = 0; j < 4; ++j)
      wf[j] = load_frag(&wtile[cur][(j * 16 + ln) * 32 + koff]);
    if (has_next)  // global_prefetch_b8 for the A side of the next K-slab
      __builtin_prefetch(A + (long)(m0 + ln) * lda + kb + 32 + koff, 0, 0);
#pragma unroll
    for (int i = 0; i < 4; ++i)
#pragma unroll
      for (int j = 0; j < 4; ++j)
        acc[i][j] = wmma_bf16(af[i], wf[j], acc[i][j]);

    __syncthreads();                   // all waves done reading cur slab
  }

#pragma unroll
  for (int i = 0; i < 4; ++i) {
#pragma unroll
    for (int j = 0; j < 4; ++j) {
#pragma unroll
      for (int vi = 0; vi < 8; ++vi) {
        // C/D layout: VGPR vi -> M = vi (lanes 0-15) or vi+8 (lanes 16-31)
        const int row = m0 + i * 16 + vi + hf * 8;
        const int col = n0 + j * 16 + ln;
        float v = acc[i][j][vi];
        if (HAS_BIAS) v += bias[col];
        if (DO_GELU)  v = gelu_exact(v);
        if (DO_RESID) v += resid[(long)row * ldc + col];
        if (TRANS_V) {
          // Vt[(b*HEADS + head)*HS + col, s], ld = SEQ (head = z)
          const int bb = row >> 11;            // row / SEQ
          const int s  = row & (SEQ - 1);
          Cb[((long)((bb * HEADS + z) * HS + col)) * SEQ + s] = (bf16)v;
        } else if (OUT_BF16) {
          Cb[(long)row * ldc + col] = (bf16)v;
        } else {
          Cf[(long)row * ldc + col] = v;
        }
      }
    }
  }
}

// ---------------------------------------------------------------------------
// Flash attention: one wave per (b, h, 16-row query tile). K processed in
// 32-key slabs: 4 WMMAs for Q*K^T, online softmax (16-lane shfl reductions),
// probs staged through LDS to re-layout C->A fragments, 4 WMMAs for P*V
// against the pre-transposed Vt.
// ---------------------------------------------------------------------------
__global__ __launch_bounds__(128)
void flash_attn_ker(const bf16* __restrict__ Q, const bf16* __restrict__ Km,
                    const bf16* __restrict__ Vt, bf16* __restrict__ O)
{
  __shared__ __align__(16) bf16 lds_p[4][16 * 32];

  const int wave = threadIdx.x >> 5;
  const int lane = threadIdx.x & 31;
  const int ln   = lane & 15;
  const int hf   = lane >> 4;
  const int koff = hf * 8;

  const int gw = blockIdx.x * 4 + wave;
  const int qt = gw & (SEQ / 16 - 1);
  const int h  = (gw >> 7) & (HEADS - 1);
  const int b  = gw >> 11;
  const int qb = qt * 16;
  const float scale = 0.125f;  // 1/sqrt(HS)

  const bf16* qbase = Q + (long)(b * SEQ + qb + ln) * EMB + h * HS + koff;
  const v16bf qf0 = load_frag(qbase);
  const v16bf qf1 = load_frag(qbase + 32);

  float mrow[8], lrow[8];
  v8f o0 = {}, o1 = {}, o2 = {}, o3 = {};
#pragma unroll
  for (int i = 0; i < 8; ++i) { mrow[i] = -1e30f; lrow[i] = 0.0f; }

  bf16* lp = lds_p[wave];

  for (int kt = 0; kt < qb + 16; kt += 32) {
    v8f s0 = {}, s1 = {};
    {
      const bf16* kb0 = Km + (long)(b * SEQ + kt + ln) * EMB + h * HS + koff;
      s0 = wmma_bf16(qf0, load_frag(kb0), s0);
      s0 = wmma_bf16(qf1, load_frag(kb0 + 32), s0);
      const bf16* kb1 = kb0 + (long)16 * EMB;
      s1 = wmma_bf16(qf0, load_frag(kb1), s1);
      s1 = wmma_bf16(qf1, load_frag(kb1 + 32), s1);
    }
#pragma unroll
    for (int vi = 0; vi < 8; ++vi) {
      const int qrow = qb + vi + hf * 8;
      float x0 = s0[vi] * scale; if (kt + ln > qrow)      x0 = -1e30f;
      float x1 = s1[vi] * scale; if (kt + 16 + ln > qrow) x1 = -1e30f;
      // row max across the 16 lanes that hold this row's columns
      float t = fmaxf(x0, x1);
      t = fmaxf(t, __shfl_xor(t, 8, 16));
      t = fmaxf(t, __shfl_xor(t, 4, 16));
      t = fmaxf(t, __shfl_xor(t, 2, 16));
      t = fmaxf(t, __shfl_xor(t, 1, 16));
      const float mnew  = fmaxf(mrow[vi], t);
      const float alpha = __expf(mrow[vi] - mnew);
      mrow[vi] = mnew;
      const float p0 = __expf(x0 - mnew);
      const float p1 = __expf(x1 - mnew);
      const int row = vi + hf * 8;
      lp[row * 32 + ln]      = (bf16)p0;
      lp[row * 32 + 16 + ln] = (bf16)p1;
      float rs = p0 + p1;
      rs += __shfl_xor(rs, 8, 16);
      rs += __shfl_xor(rs, 4, 16);
      rs += __shfl_xor(rs, 2, 16);
      rs += __shfl_xor(rs, 1, 16);
      lrow[vi] = lrow[vi] * alpha + rs;
      o0[vi] *= alpha; o1[vi] *= alpha; o2[vi] *= alpha; o3[vi] *= alpha;
    }
    // ensure our wave's LDS prob stores landed before fragment reload
    asm volatile("s_wait_dscnt 0" ::: "memory");
    const v16bf pf = load_frag(lp + ln * 32 + koff);
    const bf16* vb = Vt + (long)((b * HEADS + h) * HS + ln) * SEQ + kt + koff;
    o0 = wmma_bf16(pf, load_frag(vb), o0);
    o1 = wmma_bf16(pf, load_frag(vb + (long)16 * SEQ), o1);
    o2 = wmma_bf16(pf, load_frag(vb + (long)32 * SEQ), o2);
    o3 = wmma_bf16(pf, load_frag(vb + (long)48 * SEQ), o3);
  }

#pragma unroll
  for (int vi = 0; vi < 8; ++vi) {
    const long orow = (long)(b * SEQ + qb + vi + hf * 8) * EMB + h * HS;
    const float inv = 1.0f / lrow[vi];
    O[orow + 0  + ln] = (bf16)(o0[vi] * inv);
    O[orow + 16 + ln] = (bf16)(o1[vi] * inv);
    O[orow + 32 + ln] = (bf16)(o2[vi] * inv);
    O[orow + 48 + ln] = (bf16)(o3[vi] * inv);
  }
}

// ---------------------------------------------------------------------------
// LayerNorm over EMB=1024: one block per row, 128 threads * 8 elems, bf16 out.
// ---------------------------------------------------------------------------
__global__ __launch_bounds__(128)
void layernorm_ker(const float* __restrict__ x, const float* __restrict__ g,
                   const float* __restrict__ bta, bf16* __restrict__ out)
{
  const int row = blockIdx.x;
  const int t = threadIdx.x;
  const float* xr = x + (long)row * EMB;
  float v[8];
  float s = 0.0f, s2 = 0.0f;
#pragma unroll
  for (int i = 0; i < 8; ++i) {
    const float f = xr[t + i * 128];
    v[i] = f; s += f; s2 += f * f;
  }
#pragma unroll
  for (int off = 16; off >= 1; off >>= 1) {
    s  += __shfl_xor(s,  off, 32);
    s2 += __shfl_xor(s2, off, 32);
  }
  __shared__ float sh[8];
  const int wv = t >> 5, lane = t & 31;
  if (lane == 0) { sh[wv] = s; sh[4 + wv] = s2; }
  __syncthreads();
  s  = sh[0] + sh[1] + sh[2] + sh[3];
  s2 = sh[4] + sh[5] + sh[6] + sh[7];
  const float mean = s * (1.0f / EMB);
  const float var  = s2 * (1.0f / EMB) - mean * mean;
  const float rstd = rsqrtf(var + 1e-5f);
#pragma unroll
  for (int i = 0; i < 8; ++i) {
    const int c = t + i * 128;
    out[(long)row * EMB + c] = (bf16)((v[i] - mean) * rstd * g[c] + bta[c]);
  }
}

// ---------------------------------------------------------------------------
__global__ void embed_ker(const int* __restrict__ tokens,
                          const float* __restrict__ emb,
                          const float* __restrict__ pos,
                          float* __restrict__ x)
{
  const long i = (long)blockIdx.x * 256 + threadIdx.x;
  if (i >= (long)NTOK * EMB) return;
  const int  e  = (int)(i & (EMB - 1));
  const long bs = i >> 10;
  const int  s  = (int)(bs & (SEQ - 1));
  const int  tok = tokens[bs];
  x[i] = emb[(long)tok * EMB + e] + pos[(long)s * EMB + e];
}

__global__ void f32_to_bf16_ker(const float* __restrict__ s,
                                bf16* __restrict__ d, long n)
{
  for (long i = (long)blockIdx.x * 256 + threadIdx.x; i < n;
       i += (long)gridDim.x * 256)
    d[i] = (bf16)s[i];
}

// ---------------------------------------------------------------------------
extern "C" void kernel_launch(void* const* d_in, const int* in_sizes, int n_in,
                              void* d_out, int out_size, void* d_ws, size_t ws_size,
                              hipStream_t stream)
{
  (void)in_sizes; (void)n_in; (void)out_size; (void)ws_size;
  const int*   tokens = (const int*)  d_in[0];
  const float* emb    = (const float*)d_in[1];
  const float* pos    = (const float*)d_in[2];
  const float* ln1_g  = (const float*)d_in[3];
  const float* ln1_b  = (const float*)d_in[4];
  const float* Wq     = (const float*)d_in[5];
  const float* Wk     = (const float*)d_in[6];
  const float* Wv     = (const float*)d_in[7];
  const float* Wo     = (const float*)d_in[8];
  const float* ln2_g  = (const float*)d_in[9];
  const float* ln2_b  = (const float*)d_in[10];
  const float* W1     = (const float*)d_in[11];
  const float* b1     = (const float*)d_in[12];
  const float* W2     = (const float*)d_in[13];
  const float* b2     = (const float*)d_in[14];
  const float* lnf_g  = (const float*)d_in[15];
  const float* lnf_b  = (const float*)d_in[16];
  const float* lm_W   = (const float*)d_in[17];
  const float* lm_b   = (const float*)d_in[18];
  float* out = (float*)d_out;

  char* ws = (char*)d_ws;
  size_t cur = 0;
  auto alloc = [&](size_t bytes) -> void* {
    void* p = ws + cur;
    cur = (cur + bytes + 255) & ~(size_t)255;
    return p;
  };
  float* x    = (float*)alloc((size_t)NTOK * EMB * 4);          // residual stream
  bf16*  xb   = (bf16*) alloc((size_t)NTOK * EMB * 2);          // LN output (GEMM A)
  bf16*  qbuf = (bf16*) alloc((size_t)NTOK * EMB * 2);
  bf16*  kbuf = (bf16*) alloc((size_t)NTOK * EMB * 2);
  bf16*  vt   = (bf16*) alloc((size_t)NTOK * EMB * 2);          // V transposed
  bf16*  ao   = (bf16*) alloc((size_t)NTOK * EMB * 2);          // attention out
  bf16*  h1b  = (bf16*) alloc((size_t)NTOK * 4 * EMB * 2);      // FFN hidden
  bf16*  wqb  = (bf16*) alloc((size_t)HEADS * HS * HS * 2);
  bf16*  wkb  = (bf16*) alloc((size_t)HEADS * HS * HS * 2);
  bf16*  wvb  = (bf16*) alloc((size_t)HEADS * HS * HS * 2);
  bf16*  wob  = (bf16*) alloc((size_t)EMB * EMB * 2);
  bf16*  w1b  = (bf16*) alloc((size_t)4 * EMB * EMB * 2);
  bf16*  w2b  = (bf16*) alloc((size_t)4 * EMB * EMB * 2);
  bf16*  lmwb = (bf16*) alloc((size_t)VOCAB * EMB * 2);

  auto conv = [&](const float* s, bf16* d, long n) {
    int blocks = (int)((n + 255) / 256);
    if (blocks > 32768) blocks = 32768;
    f32_to_bf16_ker<<<blocks, 256, 0, stream>>>(s, d, n);
  };

  // token embedding + positional encoding
  embed_ker<<<(NTOK * EMB) / 256, 256, 0, stream>>>(tokens, emb, pos, x);

  // LM head weight converted once
  conv(lm_W, lmwb, (long)VOCAB * EMB);

  const dim3 blk(128);
  const long HW = (long)HS * HS;          // per-head weight elems
  for (int l = 0; l < L_BLOCKS; ++l) {
    // ---- attention ----
    layernorm_ker<<<NTOK, blk, 0, stream>>>(x, ln1_g + (long)l * EMB,
                                            ln1_b + (long)l * EMB, xb);
    conv(Wq + (long)l * HEADS * HW, wqb, HEADS * HW);
    conv(Wk + (long)l * HEADS * HW, wkb, HEADS * HW);
    conv(Wv + (long)l * HEADS * HW, wvb, HEADS * HW);
    // per-head projections, batched over heads in blockIdx.z
    gemm_ker<1><<<dim3(16, 1, HEADS), blk, 0, stream>>>(
        xb, EMB, wqb, HS, nullptr, nullptr, qbuf, EMB, nullptr,
        NTOK, HS, HS, /*a_zoff=*/HS, /*w_zoff=*/HW, /*c_zoff=*/HS);
    gemm_ker<1><<<dim3(16, 1, HEADS), blk, 0, stream>>>(
        xb, EMB, wkb, HS, nullptr, nullptr, kbuf, EMB, nullptr,
        NTOK, HS, HS, HS, HW, HS);
    gemm_ker<17><<<dim3(16, 1, HEADS), blk, 0, stream>>>(   // V -> transposed
        xb, EMB, wvb, HS, nullptr, nullptr, vt, SEQ, nullptr,
        NTOK, HS, HS, HS, HW, 0);
    flash_attn_ker<<<(BATCH * HEADS * (SEQ / 16)) / 4, blk, 0, stream>>>(
        qbuf, kbuf, vt, ao);
    conv(Wo + (long)l * EMB * EMB, wob, (long)EMB * EMB);
    gemm_ker<8><<<dim3(16, EMB / 64, 1), blk, 0, stream>>>(  // x += ao @ Wo^T
        ao, EMB, wob, EMB, nullptr, x, nullptr, EMB, x,
        NTOK, EMB, EMB, 0, 0, 0);
    // ---- FFN ----
    layernorm_ker<<<NTOK, blk, 0, stream>>>(x, ln2_g + (long)l * EMB,
                                            ln2_b + (long)l * EMB, xb);
    conv(W1 + (long)l * 4 * EMB * EMB, w1b, (long)4 * EMB * EMB);
    gemm_ker<7><<<dim3(16, (4 * EMB) / 64, 1), blk, 0, stream>>>(  // gelu(h@W1^T+b1)
        xb, EMB, w1b, EMB, b1 + (long)l * 4 * EMB, nullptr, h1b, 4 * EMB, nullptr,
        NTOK, 4 * EMB, EMB, 0, 0, 0);
    conv(W2 + (long)l * 4 * EMB * EMB, w2b, (long)4 * EMB * EMB);
    gemm_ker<10><<<dim3(16, EMB / 64, 1), blk, 0, stream>>>(  // x += ff@W2^T+b2
        h1b, 4 * EMB, w2b, 4 * EMB, b2 + (long)l * EMB, x, nullptr, EMB, x,
        NTOK, EMB, 4 * EMB, 0, 0, 0);
  }

  // final LN + LM head
  layernorm_ker<<<NTOK, blk, 0, stream>>>(x, lnf_g, lnf_b, xb);
  gemm_ker<2><<<dim3(16, VOCAB / 64, 1), blk, 0, stream>>>(
      xb, EMB, lmwb, EMB, lm_b, out, nullptr, VOCAB, nullptr,
      NTOK, VOCAB, EMB, 0, 0, 0);
}
